// GATA_85323820302755
// MI455X (gfx1250) — compile-verified, matrix operand order checked by reading, not run on hardware
//
#include <hip/hip_runtime.h>
#include <hip/hip_bf16.h>
#include <math.h>

// ---------------------------------------------------------------------------
// Problem constants
// ---------------------------------------------------------------------------
#define NN   10000
#define EE   320000
#define DM   128
#define HH   8
#define ETILES (EE / 16)          // 20000
#define NTILES_N ((NN + 15) / 16) // 625
#define NB_RED 512                // reduction blocks

#ifndef __has_builtin
#define __has_builtin(x) 0
#endif
#if defined(__AMDGCN__) && \
    __has_builtin(__builtin_amdgcn_global_load_async_to_lds_b128) && \
    __has_builtin(__builtin_amdgcn_s_wait_asynccnt)
#define USE_ASYNC 1
#else
#define USE_ASYNC 0
#endif

#define GLOBAL_AS __attribute__((address_space(1)))
#define LDS_AS    __attribute__((address_space(3)))

typedef __attribute__((ext_vector_type(16))) __bf16 v16bf;
typedef __attribute__((ext_vector_type(8)))  float  v8f;
typedef int v4i __attribute__((vector_size(4 * sizeof(int))));

struct Frag {
    union { unsigned u[8]; v16bf v; };
};

// ---------------------------------------------------------------------------
// Workspace layout (in dwords)
// ---------------------------------------------------------------------------
#define FR_WQ   0u
#define FR_WK   8192u           // 4kt*8nt*32*8
#define FR_WV   16384u
#define FR_WO   24576u
#define FR_WE2  32768u
#define FR_WG   40960u          // 4kt*1nt*32*8 = 1024
#define FR_WE1  41984u          // 12kt*8nt*32*8 = 24576
#define FR_END  66560u
#define V_OFF     66560u                      // bf16 v: E*128 ushorts
#define LOG_OFF   (V_OFF + 20480000u)         // logits: E*8 f32
#define HUPD_OFF  (LOG_OFF + 2560000u)        // h_update: N*128 f32
#define PMAX_OFF  (HUPD_OFF + 1280000u)       // NB_RED*8
#define PSUM_OFF  (PMAX_OFF + 4096u)
#define HMAX_OFF  (PSUM_OFF + 4096u)
#define HSUM_OFF  (HMAX_OFF + 8u)

// ---------------------------------------------------------------------------
// Helpers
// ---------------------------------------------------------------------------
__device__ __forceinline__ unsigned short f32_to_bf16(float f) {
    unsigned u = __float_as_uint(f);
    u += 0x7FFFu + ((u >> 16) & 1u);           // round to nearest even
    return (unsigned short)(u >> 16);
}
__device__ __forceinline__ unsigned pack2bf(float a, float b) {
    return (unsigned)f32_to_bf16(a) | ((unsigned)f32_to_bf16(b) << 16);
}
__device__ __forceinline__ float bf16_to_f32(unsigned short h) {
    return __uint_as_float(((unsigned)h) << 16);
}

// async global->LDS copy of one b128 per lane (fp32 payload)
__device__ __forceinline__ void async_row_b128(const float* g, unsigned* l) {
#if USE_ASYNC
    __builtin_amdgcn_global_load_async_to_lds_b128(
        (GLOBAL_AS v4i*)g, (LDS_AS v4i*)l, 0, 0);
#else
    (void)g; (void)l;
#endif
}
__device__ __forceinline__ void wait_async() {
#if USE_ASYNC
    __builtin_amdgcn_s_wait_asynccnt(0);
#endif
}

// in-place fp32 -> bf16 compaction: fp32 rows at 128-dword stride become
// bf16-packed rows at 64-dword stride (same base). Safe within one wave:
// ds-loads of a row complete before its ds-stores, and the write window
// [64r, 64r+64) never touches an unconverted fp32 row (>= 128(r+1)).
__device__ __forceinline__ void compact_bf16(unsigned* buf, int lane) {
    for (int r = 0; r < 16; ++r) {
        float4 f = ((const float4*)(buf + r * 128))[lane];
        unsigned lo = pack2bf(f.x, f.y);
        unsigned hi = pack2bf(f.z, f.w);
        buf[r * 64 + lane * 2]     = lo;
        buf[r * 64 + lane * 2 + 1] = hi;
    }
}

// A-fragment (16x32 bf16) from row-major bf16-packed LDS (rowstride in dwords)
__device__ __forceinline__ void load_fragA(const unsigned* base, int rowstride,
                                           int kt, int lane, Frag& f) {
    int m = lane & 15, g = lane >> 4;
    const unsigned* p = base + m * rowstride + kt * 16 + (g ? 4 : 0);
    uint4 x0 = *(const uint4*)p;
    uint4 x1 = *(const uint4*)(p + 8);
    f.u[0] = x0.x; f.u[1] = x0.y; f.u[2] = x0.z; f.u[3] = x0.w;
    f.u[4] = x1.x; f.u[5] = x1.y; f.u[6] = x1.z; f.u[7] = x1.w;
}

// B-fragment from pre-swizzled global fragment array
__device__ __forceinline__ void load_fragB(const unsigned* F, int NT,
                                           int kt, int nt, int lane, Frag& f) {
    const unsigned* p = F + ((size_t)(kt * NT + nt) * 32 + lane) * 8;
    uint4 x0 = *(const uint4*)p;
    uint4 x1 = *(const uint4*)(p + 4);
    f.u[0] = x0.x; f.u[1] = x0.y; f.u[2] = x0.z; f.u[3] = x0.w;
    f.u[4] = x1.x; f.u[5] = x1.y; f.u[6] = x1.z; f.u[7] = x1.w;
}

__device__ __forceinline__ v8f wmma_bf16(const Frag& a, const Frag& b, v8f c) {
    return __builtin_amdgcn_wmma_f32_16x16x32_bf16(
        false, a.v, false, b.v, (short)0, c, false, false);
}

// ---------------------------------------------------------------------------
// Weight repack: W [Ktot, ncols] f32 -> bf16 fragment order
// ---------------------------------------------------------------------------
__global__ void prep_frag(const float* __restrict__ W, unsigned* __restrict__ F,
                          int Ktot, int ncols, int NT) {
    int kt = blockIdx.x / NT, nt = blockIdx.x % NT;
    int t = threadIdx.x;
    int lane = t >> 3, j = t & 7;
    int nl = lane & 15, g = lane >> 4;
    int n = nt * 16 + nl;
    int k;
    if (j < 4) k = kt * 32 + (g ? 8 : 0) + 2 * j;
    else       k = kt * 32 + 16 + (g ? 8 : 0) + 2 * (j - 4);
    float v0 = 0.f, v1 = 0.f;
    if (n < ncols && k < Ktot)       v0 = W[(size_t)k * ncols + n];
    if (n < ncols && (k + 1) < Ktot) v1 = W[(size_t)(k + 1) * ncols + n];
    F[((size_t)(kt * NT + nt) * 32 + lane) * 8 + j] = pack2bf(v0, v1);
}

__global__ void zero_f32(float* p, int n) {
    int i = blockIdx.x * blockDim.x + threadIdx.x;
    for (; i < n; i += gridDim.x * blockDim.x) p[i] = 0.f;
}

// ---------------------------------------------------------------------------
// Kernel 1: gather + q/k/v GEMMs + per-head logits (q·k/4 + t_ij@Wg + bg)
// 2 waves/block, one 16-edge tile per wave; async gather into LDS
// ---------------------------------------------------------------------------
__global__ __launch_bounds__(64) void qkv_logits(
    const int* __restrict__ ei, const float* __restrict__ h,
    const float* __restrict__ tij,
    const unsigned* __restrict__ WqF, const unsigned* __restrict__ WkF,
    const unsigned* __restrict__ WvF, const unsigned* __restrict__ WgF,
    const float* __restrict__ bq, const float* __restrict__ bk,
    const float* __restrict__ bv, const float* __restrict__ bg,
    unsigned short* __restrict__ v_ws, float* __restrict__ logits) {
    __shared__ unsigned lds[2 * 3 * 2048];  // per wave: 3 x 2048 dw staging
    __shared__ float lt[2][128];            // per wave 16x8 logit tile

    int wave = threadIdx.x >> 5, lane = threadIdx.x & 31;
    int tile = blockIdx.x * 2 + wave;
    if (tile >= ETILES) return;
    int e0 = tile * 16;
    unsigned* Asrc = lds + wave * 6144;
    unsigned* Adst = Asrc + 2048;
    unsigned* At   = Adst + 2048;

    __builtin_prefetch(WqF, 0, 2);
    __builtin_prefetch(WkF, 0, 2);
    __builtin_prefetch(WvF, 0, 2);

#if USE_ASYNC
    for (int r = 0; r < 16; ++r) {
        int s = ei[e0 + r];
        int d = ei[EE + e0 + r];
        async_row_b128(h + (size_t)s * DM + lane * 4, Asrc + r * 128 + lane * 4);
        async_row_b128(h + (size_t)d * DM + lane * 4, Adst + r * 128 + lane * 4);
        async_row_b128(tij + (size_t)(e0 + r) * DM + lane * 4, At + r * 128 + lane * 4);
    }
    wait_async();
    compact_bf16(Asrc, lane);
    compact_bf16(Adst, lane);
    compact_bf16(At, lane);
#else
    for (int r = 0; r < 16; ++r) {
        int s = ei[e0 + r];
        int d = ei[EE + e0 + r];
        float4 fs = ((const float4*)(h + (size_t)s * DM))[lane];
        float4 fd = ((const float4*)(h + (size_t)d * DM))[lane];
        float4 ft = ((const float4*)(tij + (size_t)(e0 + r) * DM))[lane];
        Asrc[r * 64 + lane * 2]     = pack2bf(fs.x, fs.y);
        Asrc[r * 64 + lane * 2 + 1] = pack2bf(fs.z, fs.w);
        Adst[r * 64 + lane * 2]     = pack2bf(fd.x, fd.y);
        Adst[r * 64 + lane * 2 + 1] = pack2bf(fd.z, fd.w);
        At[r * 64 + lane * 2]       = pack2bf(ft.x, ft.y);
        At[r * 64 + lane * 2 + 1]   = pack2bf(ft.z, ft.w);
    }
#endif

    int col = lane & 15, g = lane >> 4;

    for (int nt = 0; nt < HH; ++nt) {
        v8f aq = {}, ak = {}, av = {};
        for (int kt = 0; kt < 4; ++kt) {
            Frag aS, aD, bQ, bK, bV;
            load_fragA(Adst, 64, kt, lane, aD);
            load_fragA(Asrc, 64, kt, lane, aS);
            load_fragB(WqF, 8, kt, nt, lane, bQ);
            load_fragB(WkF, 8, kt, nt, lane, bK);
            load_fragB(WvF, 8, kt, nt, lane, bV);
            aq = wmma_bf16(aD, bQ, aq);
            ak = wmma_bf16(aS, bK, ak);
            av = wmma_bf16(aS, bV, av);
        }
        float bqv = bq[nt * 16 + col];
        float bkv = bk[nt * 16 + col];
        float bvv = bv[nt * 16 + col];
        #pragma unroll
        for (int i = 0; i < 8; ++i) { aq[i] += bqv; ak[i] += bkv; av[i] += bvv; }
        #pragma unroll
        for (int r = 0; r < 8; ++r) {
            int row = r + 8 * g;
            v_ws[(size_t)(e0 + row) * DM + nt * 16 + col] = f32_to_bf16(av[r]);
        }
        #pragma unroll
        for (int r = 0; r < 8; ++r) {
            float t = aq[r] * ak[r];
            t += __shfl_xor(t, 1, 32);
            t += __shfl_xor(t, 2, 32);
            t += __shfl_xor(t, 4, 32);
            t += __shfl_xor(t, 8, 32);
            if (col == 0) lt[wave][(r + 8 * g) * 8 + nt] = t * 0.25f;
        }
    }
    // gate: t_ij @ Wg (padded to 16 cols) + bg
    v8f ag = {};
    for (int kt = 0; kt < 4; ++kt) {
        Frag aT, bG;
        load_fragA(At, 64, kt, lane, aT);
        load_fragB(WgF, 1, kt, 0, lane, bG);
        ag = wmma_bf16(aT, bG, ag);
    }
    if (col < HH) {
        float bgv = bg[col];
        #pragma unroll
        for (int r = 0; r < 8; ++r) lt[wave][(r + 8 * g) * 8 + col] += ag[r] + bgv;
    }
    float4 lv = ((const float4*)lt[wave])[lane];
    ((float4*)(logits + (size_t)e0 * HH))[lane] = lv;
}

// ---------------------------------------------------------------------------
// Global per-head softmax reductions (deterministic tree)
// ---------------------------------------------------------------------------
__global__ void red_max(const float* __restrict__ logits, float* __restrict__ pmax) {
    __shared__ float sm[256 * 8];
    float m[8];
    #pragma unroll
    for (int hh = 0; hh < 8; ++hh) m[hh] = -3.4e38f;
    for (int e = blockIdx.x * blockDim.x + threadIdx.x; e < EE;
         e += gridDim.x * blockDim.x) {
        const float4* lp = (const float4*)(logits + (size_t)e * 8);
        float4 a = lp[0], b = lp[1];
        m[0] = fmaxf(m[0], a.x); m[1] = fmaxf(m[1], a.y);
        m[2] = fmaxf(m[2], a.z); m[3] = fmaxf(m[3], a.w);
        m[4] = fmaxf(m[4], b.x); m[5] = fmaxf(m[5], b.y);
        m[6] = fmaxf(m[6], b.z); m[7] = fmaxf(m[7], b.w);
    }
    #pragma unroll
    for (int hh = 0; hh < 8; ++hh) sm[threadIdx.x * 8 + hh] = m[hh];
    __syncthreads();
    if (threadIdx.x < 8) {
        float mm = -3.4e38f;
        for (int t = 0; t < 256; ++t) mm = fmaxf(mm, sm[t * 8 + threadIdx.x]);
        pmax[blockIdx.x * 8 + threadIdx.x] = mm;
    }
}
__global__ void final_max(const float* __restrict__ pmax, float* __restrict__ hmax) {
    if (threadIdx.x < 8) {
        float mm = -3.4e38f;
        for (int b = 0; b < NB_RED; ++b) mm = fmaxf(mm, pmax[b * 8 + threadIdx.x]);
        hmax[threadIdx.x] = mm;
    }
}
__global__ void red_sum(const float* __restrict__ logits,
                        const float* __restrict__ hmax, float* __restrict__ psum) {
    __shared__ float sm[256 * 8];
    float mx[8], s[8];
    #pragma unroll
    for (int hh = 0; hh < 8; ++hh) { mx[hh] = hmax[hh]; s[hh] = 0.f; }
    for (int e = blockIdx.x * blockDim.x + threadIdx.x; e < EE;
         e += gridDim.x * blockDim.x) {
        const float4* lp = (const float4*)(logits + (size_t)e * 8);
        float4 a = lp[0], b = lp[1];
        s[0] += __expf(a.x - mx[0]); s[1] += __expf(a.y - mx[1]);
        s[2] += __expf(a.z - mx[2]); s[3] += __expf(a.w - mx[3]);
        s[4] += __expf(b.x - mx[4]); s[5] += __expf(b.y - mx[5]);
        s[6] += __expf(b.z - mx[6]); s[7] += __expf(b.w - mx[7]);
    }
    #pragma unroll
    for (int hh = 0; hh < 8; ++hh) sm[threadIdx.x * 8 + hh] = s[hh];
    __syncthreads();
    if (threadIdx.x < 8) {
        float ss = 0.f;
        for (int t = 0; t < 256; ++t) ss += sm[t * 8 + threadIdx.x];
        psum[blockIdx.x * 8 + threadIdx.x] = ss;
    }
}
__global__ void final_sum(const float* __restrict__ psum, float* __restrict__ hsum) {
    if (threadIdx.x < 8) {
        float ss = 0.f;
        for (int b = 0; b < NB_RED; ++b) ss += psum[b * 8 + threadIdx.x];
        hsum[threadIdx.x] = ss;
    }
}

// ---------------------------------------------------------------------------
// Scatter-add of alpha * v into h_update[dst]
// ---------------------------------------------------------------------------
__global__ void scatter_msg(const int* __restrict__ ei,
                            const unsigned short* __restrict__ v_ws,
                            const float* __restrict__ logits,
                            const float* __restrict__ hmax,
                            const float* __restrict__ hsum,
                            float* __restrict__ h_update) {
    size_t idx = (size_t)blockIdx.x * blockDim.x + threadIdx.x;
    size_t total = (size_t)EE * DM;
    for (; idx < total; idx += (size_t)gridDim.x * blockDim.x) {
        int e = (int)(idx >> 7), d = (int)(idx & 127);
        int hh = d >> 4;
        float l = logits[(size_t)e * HH + hh];
        float a = __expf(l - hmax[hh]) / hsum[hh];
        float vv = bf16_to_f32(v_ws[idx]);
        int dst = ei[EE + e];
        atomicAdd(&h_update[(size_t)dst * DM + d], a * vv);
    }
}

// ---------------------------------------------------------------------------
// h_new = h + h_update @ Wo + bo  ->  d_out[0 : N*D]
// ---------------------------------------------------------------------------
__global__ __launch_bounds__(128) void hnew_kernel(
    const float* __restrict__ h, const float* __restrict__ h_update,
    const unsigned* __restrict__ WoF, const float* __restrict__ bo,
    float* __restrict__ out) {
    __shared__ unsigned lds[4 * 2048];
    int wave = threadIdx.x >> 5, lane = threadIdx.x & 31;
    int tile = blockIdx.x * 4 + wave;
    if (tile >= NTILES_N) return;
    int n0 = tile * 16;
    unsigned* A = lds + wave * 2048;
    __builtin_prefetch(WoF, 0, 2);
#if USE_ASYNC
    for (int r = 0; r < 16; ++r)
        async_row_b128(h_update + (size_t)(n0 + r) * DM + lane * 4,
                       A + r * 128 + lane * 4);
    wait_async();
    compact_bf16(A, lane);
#else
    for (int r = 0; r < 16; ++r) {
        float4 f = ((const float4*)(h_update + (size_t)(n0 + r) * DM))[lane];
        A[r * 64 + lane * 2]     = pack2bf(f.x, f.y);
        A[r * 64 + lane * 2 + 1] = pack2bf(f.z, f.w);
    }
#endif
    int col = lane & 15, g = lane >> 4;
    for (int nt = 0; nt < 8; ++nt) {
        v8f acc = {};
        for (int kt = 0; kt < 4; ++kt) {
            Frag a, b;
            load_fragA(A, 64, kt, lane, a);
            load_fragB(WoF, 8, kt, nt, lane, b);
            acc = wmma_bf16(a, b, acc);
        }
        float bov = bo[nt * 16 + col];
        #pragma unroll
        for (int r = 0; r < 8; ++r) {
            int row = r + 8 * g;
            size_t o = (size_t)(n0 + row) * DM + nt * 16 + col;
            out[o] = h[o] + acc[r] + bov;
        }
    }
}

// ---------------------------------------------------------------------------
// Fused edge MLP: t_new = t_ij + silu([hn[src],hn[dst],t_ij]@We1+be1)@We2+be2
// 2 waves/block; async gather; K=384 handled as 3 LDS segments x 4 K-tiles
// ---------------------------------------------------------------------------
__global__ __launch_bounds__(64) void edge_mlp(
    const int* __restrict__ ei, const float* __restrict__ hn,
    const float* __restrict__ tij,
    const unsigned* __restrict__ We1F, const float* __restrict__ be1,
    const unsigned* __restrict__ We2F, const float* __restrict__ be2,
    float* __restrict__ outT) {
    __shared__ unsigned lds[2 * 3 * 2048];  // per wave: 3 segment buffers
    __shared__ unsigned ldsH[2 * 1024];     // per wave: hidden (bf16)
    int wave = threadIdx.x >> 5, lane = threadIdx.x & 31;
    int tile = blockIdx.x * 2 + wave;
    if (tile >= ETILES) return;
    int e0 = tile * 16;
    unsigned* S0 = lds + wave * 6144;       // hn[src]
    unsigned* S1 = S0 + 2048;               // hn[dst]
    unsigned* S2 = S1 + 2048;               // t_ij
    unsigned* Hd = ldsH + wave * 1024;

    __builtin_prefetch(We1F, 0, 2);
    __builtin_prefetch(We2F, 0, 2);

#if USE_ASYNC
    for (int r = 0; r < 16; ++r) {
        int s = ei[e0 + r];
        int d = ei[EE + e0 + r];
        async_row_b128(hn + (size_t)s * DM + lane * 4, S0 + r * 128 + lane * 4);
        async_row_b128(hn + (size_t)d * DM + lane * 4, S1 + r * 128 + lane * 4);
        async_row_b128(tij + (size_t)(e0 + r) * DM + lane * 4, S2 + r * 128 + lane * 4);
    }
    wait_async();
    compact_bf16(S0, lane);
    compact_bf16(S1, lane);
    compact_bf16(S2, lane);
#else
    for (int r = 0; r < 16; ++r) {
        int s = ei[e0 + r];
        int d = ei[EE + e0 + r];
        float4 fs = ((const float4*)(hn + (size_t)s * DM))[lane];
        float4 fd = ((const float4*)(hn + (size_t)d * DM))[lane];
        float4 ft = ((const float4*)(tij + (size_t)(e0 + r) * DM))[lane];
        S0[r * 64 + lane * 2]     = pack2bf(fs.x, fs.y);
        S0[r * 64 + lane * 2 + 1] = pack2bf(fs.z, fs.w);
        S1[r * 64 + lane * 2]     = pack2bf(fd.x, fd.y);
        S1[r * 64 + lane * 2 + 1] = pack2bf(fd.z, fd.w);
        S2[r * 64 + lane * 2]     = pack2bf(ft.x, ft.y);
        S2[r * 64 + lane * 2 + 1] = pack2bf(ft.z, ft.w);
    }
#endif
    int col = lane & 15, g = lane >> 4;
    unsigned short* hidw = (unsigned short*)Hd;
    unsigned* segs[3] = {S0, S1, S2};

    // hidden = silu(A @ We1 + be1), K = 384 = 3 segments x 4 K-tiles
    for (int nt = 0; nt < 8; ++nt) {
        v8f acc = {};
        for (int kt = 0; kt < 12; ++kt) {
            Frag a, b;
            load_fragA(segs[kt >> 2], 64, kt & 3, lane, a);
            load_fragB(We1F, 8, kt, nt, lane, b);
            acc = wmma_bf16(a, b, acc);
        }
        float b1 = be1[nt * 16 + col];
        #pragma unroll
        for (int r = 0; r < 8; ++r) {
            int row = r + 8 * g;
            float x = acc[r] + b1;
            float sl = x / (1.f + __expf(-x));  // silu
            hidw[row * DM + nt * 16 + col] = f32_to_bf16(sl);
        }
    }
    // out = hidden @ We2 + be2 + t_ij
    for (int nt = 0; nt < 8; ++nt) {
        v8f acc = {};
        for (int kt = 0; kt < 4; ++kt) {
            Frag a, b;
            load_fragA(Hd, 64, kt, lane, a);
            load_fragB(We2F, 8, kt, nt, lane, b);
            acc = wmma_bf16(a, b, acc);
        }
        float b2 = be2[nt * 16 + col];
        #pragma unroll
        for (int r = 0; r < 8; ++r) {
            int row = r + 8 * g;
            size_t o = (size_t)(e0 + row) * DM + nt * 16 + col;
            outT[o] = tij[o] + acc[r] + b2;
        }
    }
}

// ---------------------------------------------------------------------------
// Launcher
// ---------------------------------------------------------------------------
extern "C" void kernel_launch(void* const* d_in, const int* in_sizes, int n_in,
                              void* d_out, int out_size, void* d_ws, size_t ws_size,
                              hipStream_t stream) {
    const int*   ei  = (const int*)d_in[0];
    const float* h   = (const float*)d_in[1];
    const float* tij = (const float*)d_in[2];
    const float* Wq  = (const float*)d_in[3];  const float* bq = (const float*)d_in[4];
    const float* Wk  = (const float*)d_in[5];  const float* bk = (const float*)d_in[6];
    const float* Wv  = (const float*)d_in[7];  const float* bv = (const float*)d_in[8];
    const float* Wg  = (const float*)d_in[9];  const float* bg = (const float*)d_in[10];
    const float* Wo  = (const float*)d_in[11]; const float* bo = (const float*)d_in[12];
    const float* We1 = (const float*)d_in[13]; const float* be1 = (const float*)d_in[14];
    const float* We2 = (const float*)d_in[15]; const float* be2 = (const float*)d_in[16];

    unsigned* ws = (unsigned*)d_ws;
    float* out = (float*)d_out;
    unsigned short* v_ws = (unsigned short*)(ws + V_OFF);
    float* logits = (float*)(ws + LOG_OFF);
    float* h_update = (float*)(ws + HUPD_OFF);
    float* pmax = (float*)(ws + PMAX_OFF);
    float* psum = (float*)(ws + PSUM_OFF);
    float* hmax = (float*)(ws + HMAX_OFF);
    float* hsum = (float*)(ws + HSUM_OFF);

    prep_frag<<<32, 256, 0, stream>>>(Wq,  ws + FR_WQ,  128, 128, 8);
    prep_frag<<<32, 256, 0, stream>>>(Wk,  ws + FR_WK,  128, 128, 8);
    prep_frag<<<32, 256, 0, stream>>>(Wv,  ws + FR_WV,  128, 128, 8);
    prep_frag<<<32, 256, 0, stream>>>(Wo,  ws + FR_WO,  128, 128, 8);
    prep_frag<<<32, 256, 0, stream>>>(We2, ws + FR_WE2, 128, 128, 8);
    prep_frag<<< 4, 256, 0, stream>>>(Wg,  ws + FR_WG,  128,   8, 1);
    prep_frag<<<96, 256, 0, stream>>>(We1, ws + FR_WE1, 384, 128, 8);

    zero_f32<<<5000, 256, 0, stream>>>(h_update, NN * DM);

    qkv_logits<<<ETILES / 2, 64, 0, stream>>>(
        ei, h, tij, ws + FR_WQ, ws + FR_WK, ws + FR_WV, ws + FR_WG,
        bq, bk, bv, bg, v_ws, logits);

    red_max<<<NB_RED, 256, 0, stream>>>(logits, pmax);
    final_max<<<1, 64, 0, stream>>>(pmax, hmax);
    red_sum<<<NB_RED, 256, 0, stream>>>(logits, hmax, psum);
    final_sum<<<1, 64, 0, stream>>>(psum, hsum);

    scatter_msg<<<160000, 256, 0, stream>>>(ei, v_ws, logits, hmax, hsum, h_update);

    hnew_kernel<<<(NTILES_N + 3) / 4, 128, 0, stream>>>(
        h, h_update, ws + FR_WO, bo, out);

    edge_mlp<<<ETILES / 2, 64, 0, stream>>>(
        ei, out, tij, ws + FR_WE1, be1, ws + FR_WE2, be2, out + (size_t)NN * DM);
}
